// Graphormer3DEncoderLayer_84679575208544
// MI455X (gfx1250) — matrix operand order repaired and checked by reading, not compiled, not run on hardware
//
#include <hip/hip_runtime.h>
#include <hip/hip_bf16.h>
#include <math.h>

#define N_TOK 1024
#define G_DIM 4
#define E_DIM 768
#define H_DIM 8
#define F_DIM 3072
#define D_DIM 96
#define NHEAD 32          // G*H
#define MROWS 4096        // N*G

typedef _Float16 h16;
typedef __attribute__((ext_vector_type(8)))  h16   v8h;
typedef __attribute__((ext_vector_type(16))) h16   v16h;
typedef __attribute__((ext_vector_type(8)))  float v8f;
typedef __attribute__((ext_vector_type(4)))  int   v4i;

#if defined(__has_builtin)
#if __has_builtin(__builtin_amdgcn_global_load_async_to_lds_b128) && \
    __has_builtin(__builtin_amdgcn_s_wait_asynccnt)
#define USE_ASYNC 1
#endif
#endif
#ifndef USE_ASYNC
#define USE_ASYNC 0
#endif

typedef __attribute__((address_space(1))) v4i as1_v4i;
typedef __attribute__((address_space(3))) v4i as3_v4i;

// one 16B async (or sync fallback) copy global -> LDS
static __device__ inline void cp16(h16* lds, const h16* g) {
#if USE_ASYNC
  __builtin_amdgcn_global_load_async_to_lds_b128((as1_v4i*)g, (as3_v4i*)lds, 0, 0);
#else
  *(v8h*)lds = *(const v8h*)g;
#endif
}

static __device__ inline v8f wmma_f16(v16h a, v16h b, v8f c) {
  return __builtin_amdgcn_wmma_f32_16x16x32_f16(false, a, false, b, (short)0, c,
                                                false, false);
}

static __device__ inline float fast_tanh(float x) {
#if __has_builtin(__builtin_amdgcn_tanhf)
  return __builtin_amdgcn_tanhf(x);
#elif __has_builtin(__builtin_amdgcn_tanh_f32)
  return __builtin_amdgcn_tanh_f32(x);
#else
  return tanhf(x);
#endif
}

// A-fragment (16x32 f16, row-major source, K contiguous). lane l: row = l&15,
// K chunks [g*8,+8) and [16+g*8,+8) with g = l>>4.  Two 16B loads.
static __device__ inline v16h frag_a(const h16* base, int ld, int lane) {
  int m = lane & 15, g = lane >> 4;
  const h16* p = base + (size_t)m * ld + g * 8;
  v8h lo = *(const v8h*)(p);
  v8h hi = *(const v8h*)(p + 16);
  return __builtin_shufflevector(lo, hi, 0,1,2,3,4,5,6,7,8,9,10,11,12,13,14,15);
}

// B-fragment (32x16 f16). Source stored [n][k] row-major (k contiguous).
// lane l: n = l&15, K chunks [g*8,+8) and [16+g*8,+8) with g = l>>4.  Two 16B loads.
static __device__ inline v16h frag_b(const h16* base, int ld, int lane) {
  int n = lane & 15, g = lane >> 4;
  const h16* p = base + (size_t)n * ld + g * 8;
  v8h lo = *(const v8h*)(p);
  v8h hi = *(const v8h*)(p + 16);
  return __builtin_shufflevector(lo, hi, 0,1,2,3,4,5,6,7,8,9,10,11,12,13,14,15);
}

// ------------------------------------------------ fp32 -> f16 weight convert
__global__ __launch_bounds__(256)
void cvt_w_kernel(const float* __restrict__ src, h16* __restrict__ dst) {
  size_t i = ((size_t)blockIdx.x * 256 + threadIdx.x) * 8;
  const float4* s = (const float4*)(src + i);
  float4 f0 = s[0], f1 = s[1];
  v8h o = {(h16)f0.x,(h16)f0.y,(h16)f0.z,(h16)f0.w,
           (h16)f1.x,(h16)f1.y,(h16)f1.z,(h16)f1.w};
  *(v8h*)(dst + i) = o;
}

// ---------------------------------------------------------------- LayerNorm
__global__ __launch_bounds__(256)
void ln_kernel(const float* __restrict__ x, const float* __restrict__ gam,
               const float* __restrict__ bet, h16* __restrict__ out) {
  __shared__ float red[256];
  int row = blockIdx.x, t = threadIdx.x;
  size_t base = (size_t)row * E_DIM;
  float v0 = x[base + t], v1 = x[base + t + 256], v2 = x[base + t + 512];
  red[t] = v0 + v1 + v2;
  __syncthreads();
  for (int o = 128; o > 0; o >>= 1) { if (t < o) red[t] += red[t + o]; __syncthreads(); }
  float mean = red[0] * (1.0f / E_DIM);
  __syncthreads();
  float d0 = v0 - mean, d1 = v1 - mean, d2 = v2 - mean;
  red[t] = d0 * d0 + d1 * d1 + d2 * d2;
  __syncthreads();
  for (int o = 128; o > 0; o >>= 1) { if (t < o) red[t] += red[t + o]; __syncthreads(); }
  float rstd = rsqrtf(red[0] * (1.0f / E_DIM) + 1e-5f);
  out[base + t]       = (h16)(d0 * rstd * gam[t]       + bet[t]);
  out[base + t + 256] = (h16)(d1 * rstd * gam[t + 256] + bet[t + 256]);
  out[base + t + 512] = (h16)(d2 * rstd * gam[t + 512] + bet[t + 512]);
}

// ============ single-output GEMM: block 128M x 128N, wave 32x64, K-step 64 ====
// Shared mainloop; EPI 0 = qkv scatter, EPI 1 = out_proj (+bias +residual)
// Double-buffered async global->LDS staging when available.
template <int EPI>
__global__ __launch_bounds__(256)
void gemm_single(const h16* __restrict__ A, const h16* __restrict__ Wh,
                 const float* __restrict__ bias, const float* __restrict__ x,
                 h16* __restrict__ qh, h16* __restrict__ kh, h16* __restrict__ vt,
                 float* __restrict__ x_mid) {
#if USE_ASYNC
  __shared__ __align__(16) h16 ldsA[2][128 * 72];
  __shared__ __align__(16) h16 ldsW[2][128 * 72];
#else
  __shared__ __align__(16) h16 ldsA[1][128 * 72];
  __shared__ __align__(16) h16 ldsW[1][128 * 72];
#endif
  int t = threadIdx.x, lane = t & 31, w = t >> 5;
  int wm = w >> 1, wn = w & 1;                 // 4 x 2 waves
  int m0 = blockIdx.y * 128, n0 = blockIdx.x * 128;
  v8f zero = {0.f,0.f,0.f,0.f,0.f,0.f,0.f,0.f};
  v8f acc[2][4];
#pragma unroll
  for (int mf = 0; mf < 2; mf++)
#pragma unroll
    for (int nf = 0; nf < 4; nf++) acc[mf][nf] = zero;

  int sr = t >> 1, sc = (t & 1) * 32;          // staging coords (128 rows x 64)
  const h16* aRow = A  + (size_t)(m0 + sr) * E_DIM + sc;
  const h16* wRow = Wh + (size_t)(n0 + sr) * E_DIM + sc;

  auto issue = [&](int k0, int buf) {
#pragma unroll
    for (int j = 0; j < 4; j++) {
      cp16(&ldsA[buf][sr * 72 + sc + j * 8], aRow + k0 + j * 8);
      cp16(&ldsW[buf][sr * 72 + sc + j * 8], wRow + k0 + j * 8);
    }
  };

#if USE_ASYNC
  issue(0, 0);
#endif
  for (int k0 = 0, it = 0; k0 < E_DIM; k0 += 64, it++) {
    int buf = USE_ASYNC ? (it & 1) : 0;
#if USE_ASYNC
    if (k0 + 64 < E_DIM) {
      issue(k0 + 64, buf ^ 1);
      __builtin_amdgcn_s_wait_asynccnt(8);     // previous tile complete
    } else {
      __builtin_amdgcn_s_wait_asynccnt(0);
    }
#else
    issue(k0, 0);
    if (k0 + 64 < E_DIM) {
      __builtin_prefetch(aRow + k0 + 64, 0, 3);
      __builtin_prefetch(wRow + k0 + 64, 0, 3);
    }
#endif
    __syncthreads();
#pragma unroll
    for (int ks = 0; ks < 2; ks++) {
      v16h af[2];
#pragma unroll
      for (int mf = 0; mf < 2; mf++)
        af[mf] = frag_a(&ldsA[buf][(wm * 32 + mf * 16) * 72 + ks * 32], 72, lane);
#pragma unroll
      for (int nf = 0; nf < 4; nf++) {
        v16h bf = frag_b(&ldsW[buf][(wn * 64 + nf * 16) * 72 + ks * 32], 72, lane);
#pragma unroll
        for (int mf = 0; mf < 2; mf++) acc[mf][nf] = wmma_f16(af[mf], bf, acc[mf][nf]);
      }
    }
    __syncthreads();
  }

  int cn = lane & 15, rg = (lane >> 4) * 8;
  int part = n0 / E_DIM;                        // block-uniform (128 | 768)
  int ebase = n0 - part * E_DIM;
#pragma unroll
  for (int mf = 0; mf < 2; mf++)
#pragma unroll
    for (int nf = 0; nf < 4; nf++)
#pragma unroll
      for (int v = 0; v < 8; v++) {
        int row = m0 + wm * 32 + mf * 16 + rg + v;
        int col = n0 + wn * 64 + nf * 16 + cn;
        float val = acc[mf][nf][v] + bias[col];
        if (EPI == 0) {
          int e = ebase + wn * 64 + nf * 16 + cn;   // e in [0,768)
          int hd = (e * 683) >> 16;                 // e / 96 (exact for e < 768)
          int d = e - hd * 96;
          int n = row >> 2, g = row & 3;
          int gh = g * H_DIM + hd;
          if (part == 0)
            qh[((size_t)gh * N_TOK + n) * D_DIM + d] = (h16)(val * 0.10206207f);
          else if (part == 1)
            kh[((size_t)gh * N_TOK + n) * D_DIM + d] = (h16)val;
          else
            vt[((size_t)gh * D_DIM + d) * N_TOK + n] = (h16)val;
        } else {
          size_t idx = (size_t)row * E_DIM + col;
          x_mid[idx] = val + x[idx];
        }
      }
}

// -------------------------------------------------------- flash-style attention
// probs = tanh(gw) * softmax(q k^T + bias);  O = probs @ v
// bias is added in Phase B where columns are thread-contiguous (float4 loads).
__global__ __launch_bounds__(256)
void attn_kernel(const h16* __restrict__ qh, const h16* __restrict__ kh,
                 const h16* __restrict__ vt, const float* __restrict__ abias,
                 const float* __restrict__ gw, h16* __restrict__ attn_out) {
  __shared__ __align__(16) float ldsS[32 * 132];
  __shared__ __align__(16) h16   ldsP[32 * 136];
  __shared__ float m_run[32], l_run[32], rrow[32];
  int t = threadIdx.x, lane = t & 31, w = t >> 5;
  int b = blockIdx.y, row0 = blockIdx.x * 32;
  int gidx = b >> 3, hidx = b & 7;
  int cn = lane & 15, rg = (lane >> 4) * 8;
  if (t < 32) { m_run[t] = -1e30f; l_run[t] = 0.f; }

  v16h qf[2][3];
  {
    const h16* qbase = qh + ((size_t)b * N_TOK + row0) * D_DIM;
#pragma unroll
    for (int mf = 0; mf < 2; mf++)
#pragma unroll
      for (int ks = 0; ks < 3; ks++)
        qf[mf][ks] = frag_a(qbase + (size_t)(mf * 16) * D_DIM + ks * 32, D_DIM, lane);
  }
  v8f zero = {0.f,0.f,0.f,0.f,0.f,0.f,0.f,0.f};
  v8f oacc[2]; oacc[0] = zero; oacc[1] = zero;
  __syncthreads();

  for (int ct = 0; ct < 8; ct++) {
    int col0 = ct * 128;
    // ---- Phase A: S = Q K^T (each wave owns a 16-col slice)
    v8f sf[2]; sf[0] = zero; sf[1] = zero;
    const h16* kbase = kh + ((size_t)b * N_TOK + col0 + w * 16) * D_DIM;
#pragma unroll
    for (int ks = 0; ks < 3; ks++) {
      v16h bf = frag_b(kbase + ks * 32, D_DIM, lane);
#pragma unroll
      for (int mf = 0; mf < 2; mf++) sf[mf] = wmma_f16(qf[mf][ks], bf, sf[mf]);
    }
#pragma unroll
    for (int mf = 0; mf < 2; mf++)
#pragma unroll
      for (int v = 0; v < 8; v++)
        ldsS[(mf * 16 + rg + v) * 132 + w * 16 + cn] = sf[mf][v];
    __syncthreads();
    // ---- Phase B: +bias, online softmax, tanh(gw) weighting -> P (f16)
    {
      int row = t >> 3, c = t & 7, cs = c * 16;
      size_t rowbase = ((size_t)b * N_TOK + row0 + row) * N_TOK + col0 + cs;
      const float4* bp4 = (const float4*)(abias + rowbase);
      const float4* gp4 = (const float4*)(gw + rowbase);
      float bv[16], gv[16];
#pragma unroll
      for (int j = 0; j < 4; j++) {
        float4 bb = bp4[j], gg = gp4[j];
        bv[4*j] = bb.x; bv[4*j+1] = bb.y; bv[4*j+2] = bb.z; bv[4*j+3] = bb.w;
        gv[4*j] = gg.x; gv[4*j+1] = gg.y; gv[4*j+2] = gg.z; gv[4*j+3] = gg.w;
      }
      if (ct < 7) {           // stream prefetch of next tile (bias + gw)
        __builtin_prefetch(abias + rowbase + 128, 0, 0);
        __builtin_prefetch(gw + rowbase + 128, 0, 0);
      }
      float sv[16];
      float mx = -1e30f;
#pragma unroll
      for (int e = 0; e < 16; e++) {
        sv[e] = ldsS[row * 132 + cs + e] + bv[e];
        mx = fmaxf(mx, sv[e]);
      }
      mx = fmaxf(mx, __shfl_xor(mx, 1, 8));
      mx = fmaxf(mx, __shfl_xor(mx, 2, 8));
      mx = fmaxf(mx, __shfl_xor(mx, 4, 8));
      float mold = m_run[row];
      float mnew = fmaxf(mold, mx);
      float r = __expf(mold - mnew);
      float ps = 0.f;
#pragma unroll
      for (int e = 0; e < 16; e++) {
        float p = __expf(sv[e] - mnew);
        ps += p;
        ldsP[row * 136 + cs + e] = (h16)(fast_tanh(gv[e]) * p);
      }
      ps += __shfl_xor(ps, 1, 8);
      ps += __shfl_xor(ps, 2, 8);
      ps += __shfl_xor(ps, 4, 8);
      if (c == 0) { m_run[row] = mnew; rrow[row] = r; l_run[row] = l_run[row] * r + ps; }
    }
    __syncthreads();
    // ---- Phase C: O = O*r + P @ V  (waves 0..5 own d-slices of 16)
    if (w < 6) {
#pragma unroll
      for (int mf = 0; mf < 2; mf++)
#pragma unroll
        for (int v = 0; v < 8; v++) oacc[mf][v] *= rrow[mf * 16 + rg + v];
      const h16* vbase = vt + ((size_t)b * D_DIM + w * 16) * N_TOK + col0;
#pragma unroll
      for (int ks = 0; ks < 4; ks++) {
        v16h bfv = frag_b(vbase + ks * 32, N_TOK, lane);
#pragma unroll
        for (int mf = 0; mf < 2; mf++) {
          v16h afp = frag_a(&ldsP[(mf * 16) * 136 + ks * 32], 136, lane);
          oacc[mf] = wmma_f16(afp, bfv, oacc[mf]);
        }
      }
    }
    __syncthreads();
  }

  if (w < 6) {
#pragma unroll
    for (int mf = 0; mf < 2; mf++)
#pragma unroll
      for (int v = 0; v < 8; v++) {
        int rr = mf * 16 + rg + v;
        float val = oacc[mf][v] / l_run[rr];
        int n = row0 + rr;
        attn_out[((size_t)n * G_DIM + gidx) * E_DIM + hidx * D_DIM + w * 16 + cn] = (h16)val;
      }
  }
}

// ====== dual-slab GLU GEMM: block 64M x 128N, wave 32x32, K-step 64 ==========
// fc1: out = gelu(a * relu(gate)^2) f16;  fc2: out = resid + a*relu(gate)^2 f32
template <int KD, int GOFF, bool FC1>
__global__ __launch_bounds__(256)
void gemm_dual(const h16* __restrict__ A, const h16* __restrict__ Wh,
               const float* __restrict__ bias, const float* __restrict__ resid,
               h16* __restrict__ out_h, float* __restrict__ out_f) {
#if USE_ASYNC
  __shared__ __align__(16) h16 ldsA[2][64 * 72];
  __shared__ __align__(16) h16 ldsWa[2][128 * 72];
  __shared__ __align__(16) h16 ldsWg[2][128 * 72];
#else
  __shared__ __align__(16) h16 ldsA[1][64 * 72];
  __shared__ __align__(16) h16 ldsWa[1][128 * 72];
  __shared__ __align__(16) h16 ldsWg[1][128 * 72];
#endif
  int t = threadIdx.x, lane = t & 31, w = t >> 5;
  int wm = w >> 2, wn = w & 3;
  int m0 = blockIdx.y * 64, n0 = blockIdx.x * 128;
  v8f zero = {0.f,0.f,0.f,0.f,0.f,0.f,0.f,0.f};
  v8f accA[2][2], accG[2][2];
#pragma unroll
  for (int mf = 0; mf < 2; mf++)
#pragma unroll
    for (int nf = 0; nf < 2; nf++) { accA[mf][nf] = zero; accG[mf][nf] = zero; }

  int ar = t >> 2, ac = (t & 3) * 16;           // A staging: 64 x 64
  int wr = t >> 1, wc = (t & 1) * 32;           // W staging: 128 x 64
  const h16* aRow  = A  + (size_t)(m0 + ar) * KD + ac;
  const h16* waRow = Wh + (size_t)(n0 + wr) * KD + wc;
  const h16* wgRow = Wh + (size_t)(GOFF + n0 + wr) * KD + wc;

  auto issue = [&](int k0, int buf) {
    cp16(&ldsA[buf][ar * 72 + ac], aRow + k0);
    cp16(&ldsA[buf][ar * 72 + ac + 8], aRow + k0 + 8);
#pragma unroll
    for (int j = 0; j < 4; j++) {
      cp16(&ldsWa[buf][wr * 72 + wc + j * 8], waRow + k0 + j * 8);
      cp16(&ldsWg[buf][wr * 72 + wc + j * 8], wgRow + k0 + j * 8);
    }
  };

#if USE_ASYNC
  issue(0, 0);
#endif
  for (int k0 = 0, it = 0; k0 < KD; k0 += 64, it++) {
    int buf = USE_ASYNC ? (it & 1) : 0;
#if USE_ASYNC
    if (k0 + 64 < KD) {
      issue(k0 + 64, buf ^ 1);
      __builtin_amdgcn_s_wait_asynccnt(10);    // previous tile complete
    } else {
      __builtin_amdgcn_s_wait_asynccnt(0);
    }
#else
    issue(k0, 0);
    if (k0 + 64 < KD) {
      __builtin_prefetch(aRow + k0 + 64, 0, 3);
      __builtin_prefetch(waRow + k0 + 64, 0, 3);
      __builtin_prefetch(wgRow + k0 + 64, 0, 3);
    }
#endif
    __syncthreads();
#pragma unroll
    for (int ks = 0; ks < 2; ks++) {
      v16h af[2];
#pragma unroll
      for (int mf = 0; mf < 2; mf++)
        af[mf] = frag_a(&ldsA[buf][(wm * 32 + mf * 16) * 72 + ks * 32], 72, lane);
#pragma unroll
      for (int nf = 0; nf < 2; nf++) {
        v16h ba = frag_b(&ldsWa[buf][(wn * 32 + nf * 16) * 72 + ks * 32], 72, lane);
        v16h bg = frag_b(&ldsWg[buf][(wn * 32 + nf * 16) * 72 + ks * 32], 72, lane);
#pragma unroll
        for (int mf = 0; mf < 2; mf++) {
          accA[mf][nf] = wmma_f16(af[mf], ba, accA[mf][nf]);
          accG[mf][nf] = wmma_f16(af[mf], bg, accG[mf][nf]);
        }
      }
    }
    __syncthreads();
  }

  int cn = lane & 15, rg = (lane >> 4) * 8;
#pragma unroll
  for (int mf = 0; mf < 2; mf++)
#pragma unroll
    for (int nf = 0; nf < 2; nf++)
#pragma unroll
      for (int v = 0; v < 8; v++) {
        int row = m0 + wm * 32 + mf * 16 + rg + v;
        int col = n0 + wn * 32 + nf * 16 + cn;
        float a  = accA[mf][nf][v] + bias[col];
        float gt = accG[mf][nf][v] + bias[GOFF + col];
        float rl = fmaxf(gt, 0.f);
        float u  = a * rl * rl;                       // GLU: a * relu(gate)^2
        size_t idx = (size_t)row * GOFF + col;
        if (FC1) {
          out_h[idx] = (h16)(0.5f * u * (1.0f + erff(u * 0.70710678f)));
        } else {
          out_f[idx] = resid[idx] + u;
        }
      }
}

// ------------------------------------------------------------------- launcher
extern "C" void kernel_launch(void* const* d_in, const int* in_sizes, int n_in,
                              void* d_out, int out_size, void* d_ws, size_t ws_size,
                              hipStream_t stream) {
  (void)in_sizes; (void)n_in; (void)out_size; (void)ws_size;
  const float* x     = (const float*)d_in[0];
  const float* abias = (const float*)d_in[1];
  const float* gw    = (const float*)d_in[2];
  const float* in_w  = (const float*)d_in[3];
  const float* in_b  = (const float*)d_in[4];
  const float* out_w = (const float*)d_in[5];
  const float* out_b = (const float*)d_in[6];
  const float* ln1g  = (const float*)d_in[7];
  const float* ln1b  = (const float*)d_in[8];
  const float* fc1w  = (const float*)d_in[9];
  const float* fc1b  = (const float*)d_in[10];
  const float* fc2w  = (const float*)d_in[11];
  const float* fc2b  = (const float*)d_in[12];
  const float* ln2g  = (const float*)d_in[13];
  const float* ln2b  = (const float*)d_in[14];
  float* out = (float*)d_out;

  const size_t EH = (size_t)MROWS * E_DIM;   // 3,145,728
  char* ws = (char*)d_ws;
  h16* h1     = (h16*)ws; ws += EH * sizeof(h16);
  h16* qh     = (h16*)ws; ws += EH * sizeof(h16);
  h16* kh     = (h16*)ws; ws += EH * sizeof(h16);
  h16* vt     = (h16*)ws; ws += EH * sizeof(h16);
  h16* attn_o = (h16*)ws; ws += EH * sizeof(h16);
  h16* h2     = (h16*)ws; ws += EH * sizeof(h16);
  h16* g1     = (h16*)ws; ws += (size_t)MROWS * F_DIM * sizeof(h16);
  float* xmid = (float*)ws; ws += EH * sizeof(float);
  // f16 copies of the weight matrices (converted once per launch)
  h16* in_wh  = (h16*)ws; ws += (size_t)3 * E_DIM * E_DIM * sizeof(h16);
  h16* out_wh = (h16*)ws; ws += (size_t)E_DIM * E_DIM * sizeof(h16);
  h16* fc1_wh = (h16*)ws; ws += (size_t)2 * F_DIM * E_DIM * sizeof(h16);
  h16* fc2_wh = (h16*)ws; ws += (size_t)2 * E_DIM * F_DIM * sizeof(h16);

  cvt_w_kernel<<<(3 * E_DIM * E_DIM) / 2048, 256, 0, stream>>>(in_w,  in_wh);
  cvt_w_kernel<<<(E_DIM * E_DIM)     / 2048, 256, 0, stream>>>(out_w, out_wh);
  cvt_w_kernel<<<(2 * F_DIM * E_DIM) / 2048, 256, 0, stream>>>(fc1w,  fc1_wh);
  cvt_w_kernel<<<(2 * E_DIM * F_DIM) / 2048, 256, 0, stream>>>(fc2w,  fc2_wh);

  ln_kernel<<<MROWS, 256, 0, stream>>>(x, ln1g, ln1b, h1);
  gemm_single<0><<<dim3(18, 32), 256, 0, stream>>>(h1, in_wh, in_b, nullptr,
                                                   qh, kh, vt, nullptr);
  attn_kernel<<<dim3(32, 32), 256, 0, stream>>>(qh, kh, vt, abias, gw, attn_o);
  gemm_single<1><<<dim3(6, 32), 256, 0, stream>>>(attn_o, out_wh, out_b, x,
                                                  nullptr, nullptr, nullptr, xmid);
  ln_kernel<<<MROWS, 256, 0, stream>>>(xmid, ln2g, ln2b, h2);
  gemm_dual<E_DIM, F_DIM, true><<<dim3(24, 64), 256, 0, stream>>>(
      h2, fc1_wh, fc1b, nullptr, g1, nullptr);
  gemm_dual<F_DIM, E_DIM, false><<<dim3(6, 64), 256, 0, stream>>>(
      g1, fc2_wh, fc2b, xmid, nullptr, out);
}
